// PointnetSAModuleMSG_70961449664642
// MI455X (gfx1250) — matrix operand order repaired and checked by reading, not compiled, not run on hardware
//
#include <hip/hip_runtime.h>

// ---------------------------------------------------------------------------
// PointNet++ SA-MSG forward for MI455X (gfx1250, wave32).
//  B=8, N=16384, C_IN=32, P=1024, radii {0.5,1.0}, nsample {16,32}
//  MLPs: (35->32->64) and (35->64->128); out = new_xyz (B,P,3) ++ feat (B,192,P)
// Roofline: ~18 MB input / ~6.4 MB output vs 23.3 TB/s HBM -> everything is
// L2-resident and latency-bound, EXCEPT the MLP GEMMs (6.3 GFLOP) which we
// push through v_wmma_f32_16x16x32_f16 (f16 in, f32 accumulate). dist^2 is
// never materialized (saves ~1 GB of traffic vs the reference).
// ---------------------------------------------------------------------------

#define BB   8
#define NN   16384
#define CINN 32
#define PP   1024

typedef __attribute__((ext_vector_type(16))) _Float16 v16h;
typedef __attribute__((ext_vector_type(8)))  float    v8f;

// ---------------------------------------------------------------------------
// Furthest point sampling: 1 block per batch, 1024 threads (32 waves).
// Each thread holds 16 points + running min-dist in VGPRs (no LDS staging).
// Writes new_xyz directly to d_out.
// ---------------------------------------------------------------------------
__global__ void __launch_bounds__(1024) k_fps(const float* __restrict__ xyz,
                                              float* __restrict__ newXyz) {
  const int b   = blockIdx.x;
  const int tid = threadIdx.x;
  __shared__ float rv[32];
  __shared__ int   ri[32];
  __shared__ float cX, cY, cZ;

  const float* base = xyz + (size_t)b * NN * 3;
  float px[16], py[16], pz[16], dmin[16];
#pragma unroll
  for (int j = 0; j < 16; ++j) {
    int n = tid + j * 1024;            // strided -> coalesced initial load
    px[j] = base[n * 3 + 0];
    py[j] = base[n * 3 + 1];
    pz[j] = base[n * 3 + 2];
    dmin[j] = 1e10f;
  }
  if (tid == 0) { cX = base[0]; cY = base[1]; cZ = base[2]; }
  __syncthreads();

  for (int it = 0; it < PP; ++it) {
    const float cx = cX, cy = cY, cz = cZ;   // coords of current 'far' sample
    if (tid == 0) {                          // emit sample #it (old far, per scan semantics)
      newXyz[((size_t)b * PP + it) * 3 + 0] = cx;
      newXyz[((size_t)b * PP + it) * 3 + 1] = cy;
      newXyz[((size_t)b * PP + it) * 3 + 2] = cz;
    }
    float bestv = -1.0f; int besti = 0;
#pragma unroll
    for (int j = 0; j < 16; ++j) {
      float dx = px[j] - cx, dy = py[j] - cy, dz = pz[j] - cz;
      float d  = dx * dx + dy * dy + dz * dz;
      dmin[j]  = fminf(dmin[j], d);
      if (dmin[j] > bestv) { bestv = dmin[j]; besti = tid + j * 1024; }
    }
    // wave32 argmax reduce
#pragma unroll
    for (int off = 16; off > 0; off >>= 1) {
      float ov = __shfl_down(bestv, off, 32);
      int   oi = __shfl_down(besti, off, 32);
      if (ov > bestv) { bestv = ov; besti = oi; }
    }
    if ((tid & 31) == 0) { rv[tid >> 5] = bestv; ri[tid >> 5] = besti; }
    __syncthreads();
    if (tid < 32) {
      bestv = rv[tid]; besti = ri[tid];
#pragma unroll
      for (int off = 16; off > 0; off >>= 1) {
        float ov = __shfl_down(bestv, off, 32);
        int   oi = __shfl_down(besti, off, 32);
        if (ov > bestv) { bestv = ov; besti = oi; }
      }
      if (tid == 0) {                  // fetch next centroid (L2 hit)
        cX = base[besti * 3 + 0];
        cY = base[besti * 3 + 1];
        cZ = base[besti * 3 + 2];
      }
    }
    __syncthreads();
  }
}

// ---------------------------------------------------------------------------
// Fused dist^2 + dual-radius ball query. One wave per query point.
// First-nsample-in-index-order via ballot + prefix popcount; early exit once
// both lists are full; pad empty slots with first hit (or 0).
// ---------------------------------------------------------------------------
__global__ void __launch_bounds__(256) k_ball(const float* __restrict__ xyz,
                                              const float* __restrict__ newXyz,
                                              int* __restrict__ idx0,
                                              int* __restrict__ idx1) {
  const int lane = threadIdx.x & 31;
  const int g    = blockIdx.x * 8 + (threadIdx.x >> 5);  // (b,p) flat
  const int b    = g >> 10;
  const float* base = xyz + (size_t)b * NN * 3;
  const float qx = newXyz[(size_t)g * 3 + 0];
  const float qy = newXyz[(size_t)g * 3 + 1];
  const float qz = newXyz[(size_t)g * 3 + 2];
  int* o0 = idx0 + (size_t)g * 16;
  int* o1 = idx1 + (size_t)g * 32;

  int cnt0 = 0, cnt1 = 0, first0 = 0, first1 = 0;
  const unsigned long long ltmask = (1ull << lane) - 1ull;

  for (int bs = 0; bs < NN && (cnt0 < 16 || cnt1 < 32); bs += 32) {
    const int n = bs + lane;
    if (bs + 32 < NN) __builtin_prefetch(base + (n + 32) * 3, 0, 0);  // global_prefetch_b8
    float dx = base[n * 3 + 0] - qx;
    float dy = base[n * 3 + 1] - qy;
    float dz = base[n * 3 + 2] - qz;
    float d2 = dx * dx + dy * dy + dz * dz;
    unsigned long long m1 = __ballot(d2 < 1.0f);    // radius 1.0
    unsigned long long m0 = __ballot(d2 < 0.25f);   // radius 0.5
    if (cnt1 < 32) {
      int pos = cnt1 + (int)__popcll(m1 & ltmask);
      if ((d2 < 1.0f) && pos < 32) o1[pos] = n;
      if (cnt1 == 0 && m1) first1 = bs + (int)__builtin_ctzll(m1);
      cnt1 += (int)__popcll(m1); if (cnt1 > 32) cnt1 = 32;
    }
    if (cnt0 < 16) {
      int pos = cnt0 + (int)__popcll(m0 & ltmask);
      if ((d2 < 0.25f) && pos < 16) o0[pos] = n;
      if (cnt0 == 0 && m0) first0 = bs + (int)__builtin_ctzll(m0);
      cnt0 += (int)__popcll(m0); if (cnt0 > 16) cnt0 = 16;
    }
  }
  const int f0 = (cnt0 > 0) ? first0 : 0;
  const int f1 = (cnt1 > 0) ? first1 : 0;
  for (int s = cnt0 + lane; s < 16; s += 32) o0[s] = f0;
  for (int s = cnt1 + lane; s < 32; s += 32) o1[s] = f1;
}

// ---------------------------------------------------------------------------
// WMMA fragment loaders (per CDNA5 ISA layouts, f16, 16x16x32).
// A (16xK rows = out-channels): lane l holds row (l&15); lane-half selects K
// sub-window; element pairs (2i,2i+1) are K-consecutive.
// B (Kx16 cols): lane l holds column (l&15); lane-half selects K sub-window (same order as A).
// ---------------------------------------------------------------------------
__device__ inline v16h load_a_frag(const _Float16* __restrict__ W, int ldk,
                                   int m0, int k0, int lane) {
  const int m = m0 + (lane & 15);
  const int h = lane >> 4;
  v16h a;
#pragma unroll
  for (int e = 0; e < 16; ++e) {
    int i = e >> 1;
    int k = ((i & 4) ? 16 : 0) + h * 8 + (i & 3) * 2 + (e & 1);
    a[e] = W[m * ldk + k0 + k];
  }
  return a;
}

__device__ inline v16h load_b_frag(const _Float16* __restrict__ T, int ldk,
                                   int k0, int lane) {
  const int n = lane & 15;
  const int h = lane >> 4;
  v16h bf;
#pragma unroll
  for (int e = 0; e < 16; ++e) {
    int k = ((e & 8) ? 16 : 0) + h * 8 + (e & 7);
    bf[e] = T[n * ldk + k0 + k];
  }
  return bf;
}

// ---------------------------------------------------------------------------
// Group + MLP(2 layers, folded BN + ReLU) + maxpool-over-S.
// One wave per query point; 16-column tiles (S=16 -> 1 tile, S=32 -> 2).
// K0 padded 35 -> 64 (two k-steps of 32); zero-padded weights & activations
// contribute nothing. f16 inputs, f32 accumulation via v_wmma_f32_16x16x32_f16.
// ---------------------------------------------------------------------------
template <int S, int C0, int C1>
__global__ void __launch_bounds__(256) k_mlp(const float* __restrict__ xyz,
                                             const float* __restrict__ feats,
                                             const float* __restrict__ newXyz,
                                             const int* __restrict__ bidx,
                                             const float* __restrict__ w0,
                                             const float* __restrict__ s0,
                                             const float* __restrict__ b0,
                                             const float* __restrict__ w1,
                                             const float* __restrict__ s1,
                                             const float* __restrict__ b1,
                                             float* __restrict__ out, int chOff) {
  constexpr int KP0 = 64;
  constexpr int MT0 = C0 / 16;
  constexpr int KS0 = KP0 / 32;
  constexpr int MT1 = C1 / 16;
  constexpr int KS1 = C0 / 32;
  constexpr int NT  = S / 16;

  __shared__ _Float16 sW0[C0 * KP0];
  __shared__ _Float16 sW1[C1 * C0];
  __shared__ _Float16 sB[8][16 * KP0];   // per-wave grouped tile [n][k]
  __shared__ _Float16 sA[8][16 * C0];    // per-wave layer0 output [n][m]

  const int tid  = threadIdx.x;
  const int w    = tid >> 5;
  const int lane = tid & 31;
  const int h    = lane >> 4;
  const int nc   = lane & 15;

  // stage f16 weights (K-padded) once per block
  for (int i = tid; i < C0 * KP0; i += 256) {
    int r = i / KP0, k = i % KP0;
    sW0[i] = (k < 35) ? (_Float16)w0[r * 35 + k] : (_Float16)0.0f;
  }
  for (int i = tid; i < C1 * C0; i += 256) sW1[i] = (_Float16)w1[i];
  __syncthreads();

  const int g = blockIdx.x * 8 + w;
  const int b = g >> 10;
  const int p = g & (PP - 1);
  const float cx = newXyz[(size_t)g * 3 + 0];
  const float cy = newXyz[(size_t)g * 3 + 1];
  const float cz = newXyz[(size_t)g * 3 + 2];

  v8f mx[MT1];
#pragma unroll
  for (int mt = 0; mt < MT1; ++mt) mx[mt] = {};   // post-ReLU >= 0, so 0-init ok

  for (int t = 0; t < NT; ++t) {
    // --- stage grouped input tile: 2 lanes per column, 32 K-values each ---
    {
      const int n    = lane & 15;
      const int kh   = lane >> 4;
      const int sidx = bidx[(size_t)g * S + t * 16 + n];
      const float* pxyz = xyz + ((size_t)b * NN + sidx) * 3;
      _Float16* dst = &sB[w][n * KP0 + kh * 32];
      if (kh == 0) {
        dst[0] = (_Float16)(pxyz[0] - cx);
        dst[1] = (_Float16)(pxyz[1] - cy);
        dst[2] = (_Float16)(pxyz[2] - cz);
#pragma unroll
        for (int k = 3; k < 32; ++k)
          dst[k] = (_Float16)feats[((size_t)b * CINN + (k - 3)) * NN + sidx];
      } else {
#pragma unroll
        for (int k = 0; k < 3; ++k)   // channels 29..31 land at K=32..34
          dst[k] = (_Float16)feats[((size_t)b * CINN + (29 + k)) * NN + sidx];
#pragma unroll
        for (int k = 3; k < 32; ++k) dst[k] = (_Float16)0.0f;
      }
    }
    __syncthreads();

    // --- layer 0: (C0 x 35) GEMM + BN-fold + ReLU -> sA ---
#pragma unroll
    for (int mt = 0; mt < MT0; ++mt) {
      v8f acc = {};
#pragma unroll
      for (int ks = 0; ks < KS0; ++ks) {
        v16h a  = load_a_frag(sW0, KP0, mt * 16, ks * 32, lane);
        v16h bf = load_b_frag(sB[w], KP0, ks * 32, lane);
        acc = __builtin_amdgcn_wmma_f32_16x16x32_f16(false, a, false, bf,
                                                     (short)0, acc, false, false);
      }
#pragma unroll
      for (int r = 0; r < 8; ++r) {      // C/D layout: VGPR r -> M = r + 8*half
        int m = mt * 16 + h * 8 + r;
        float v = acc[r] * s0[m] + b0[m];
        v = v > 0.0f ? v : 0.0f;
        sA[w][nc * C0 + m] = (_Float16)v;
      }
    }
    __syncthreads();

    // --- layer 1: (C1 x C0) GEMM + BN-fold + ReLU, fold into running max ---
#pragma unroll
    for (int mt = 0; mt < MT1; ++mt) {
      v8f acc = {};
#pragma unroll
      for (int ks = 0; ks < KS1; ++ks) {
        v16h a  = load_a_frag(sW1, C0, mt * 16, ks * 32, lane);
        v16h bf = load_b_frag(sA[w], C0, ks * 32, lane);
        acc = __builtin_amdgcn_wmma_f32_16x16x32_f16(false, a, false, bf,
                                                     (short)0, acc, false, false);
      }
#pragma unroll
      for (int r = 0; r < 8; ++r) {
        int m = mt * 16 + h * 8 + r;
        float v = acc[r] * s1[m] + b1[m];
        v = v > 0.0f ? v : 0.0f;
        mx[mt][r] = fmaxf(mx[mt][r], v);
      }
    }
    __syncthreads();
  }

  // --- maxpool across the 16 columns (lanes within each half-group) ---
#pragma unroll
  for (int mt = 0; mt < MT1; ++mt) {
#pragma unroll
    for (int r = 0; r < 8; ++r) {
      float v = mx[mt][r];
#pragma unroll
      for (int off = 1; off < 16; off <<= 1)
        v = fmaxf(v, __shfl_xor(v, off, 32));
      if (nc == 0) {
        int m = chOff + mt * 16 + h * 8 + r;
        out[((size_t)b * 192 + m) * PP + p] = v;
      }
    }
  }
}

// ---------------------------------------------------------------------------
extern "C" void kernel_launch(void* const* d_in, const int* in_sizes, int n_in,
                              void* d_out, int out_size, void* d_ws, size_t ws_size,
                              hipStream_t stream) {
  const float* xyz   = (const float*)d_in[0];
  const float* feats = (const float*)d_in[1];
  const float* w0_0 = (const float*)d_in[2];
  const float* s0_0 = (const float*)d_in[3];
  const float* b0_0 = (const float*)d_in[4];
  const float* w0_1 = (const float*)d_in[5];
  const float* s0_1 = (const float*)d_in[6];
  const float* b0_1 = (const float*)d_in[7];
  const float* w1_0 = (const float*)d_in[8];
  const float* s1_0 = (const float*)d_in[9];
  const float* b1_0 = (const float*)d_in[10];
  const float* w1_1 = (const float*)d_in[11];
  const float* s1_1 = (const float*)d_in[12];
  const float* b1_1 = (const float*)d_in[13];

  float* outNewXyz = (float*)d_out;                    // (B,P,3)
  float* outFeat   = outNewXyz + (size_t)BB * PP * 3;  // (B,192,P)

  int* idx0 = (int*)d_ws;                              // (B,P,16)
  int* idx1 = idx0 + (size_t)BB * PP * 16;             // (B,P,32)

  k_fps<<<BB, 1024, 0, stream>>>(xyz, outNewXyz);
  k_ball<<<(BB * PP) / 8, 256, 0, stream>>>(xyz, outNewXyz, idx0, idx1);
  k_mlp<16, 32, 64><<<(BB * PP) / 8, 256, 0, stream>>>(
      xyz, feats, outNewXyz, idx0, w0_0, s0_0, b0_0, w0_1, s0_1, b0_1, outFeat, 0);
  k_mlp<32, 64, 128><<<(BB * PP) / 8, 256, 0, stream>>>(
      xyz, feats, outNewXyz, idx1, w1_0, s1_0, b1_0, w1_1, s1_1, b1_1, outFeat, 64);
}
